// GPT_14224931684966
// MI455X (gfx1250) — matrix operand order repaired and checked by reading, not compiled, not run on hardware
//
#include <hip/hip_runtime.h>
#include <hip/hip_bf16.h>
#include <math.h>

// ---------------------------------------------------------------------------
// GPT forward for gfx1250 (CDNA5).
//  - All operands pre-converted to f16 (weights transposed to [N][K]) so the
//    GEMM inner loop is pure f16: async global->LDS b128 copies (ASYNCcnt),
//    double-buffered LDS, v_wmma_f32_16x16x32_f16 with f32 accumulation.
// ---------------------------------------------------------------------------

typedef __attribute__((ext_vector_type(16))) _Float16 v16h;
typedef __attribute__((ext_vector_type(8)))  _Float16 v8h;
typedef __attribute__((ext_vector_type(8)))  float    v8f;

union Frag16 { v16h v; v8h h[2]; };

#define B_   2
#define S_   2048
#define E_   512
#define A_   512
#define H_   8
#define L_   4
#define V_   50257
#define HD_  64
#define FF_  (4 * E_)
#define EPS_ 1e-5f

// ---------------------------------------------------------------------------
// Async 16-byte global -> LDS copy (GLOBAL_LOAD_ASYNC_TO_LDS_B128, ASYNCcnt).
// Generic LDS address low 32 bits == LDS byte offset (ISA flat aperture rule).
// ---------------------------------------------------------------------------
__device__ __forceinline__ void async_cp16(const _Float16* g, _Float16* l)
{
    const unsigned long long ga = (unsigned long long)(uintptr_t)g;
    const unsigned           la = (unsigned)(uintptr_t)l;
    asm volatile("global_load_async_to_lds_b128 %0, %1, off"
                 :: "v"(la), "v"(ga) : "memory");
}
__device__ __forceinline__ void wait_async_3()
{
    asm volatile("s_wait_asynccnt 0x3" ::: "memory");
}
__device__ __forceinline__ void wait_async_0()
{
    asm volatile("s_wait_asynccnt 0x0" ::: "memory");
}

// ---------------------------------------------------------------------------
// Tiled WMMA GEMM (f16 x f16 -> f32):
//   A[M,K] f16 row-major (lda, halves)
//   B      f16 n-major: B[n][k] contiguous in k (ldb, halves)  [all weights
//          pre-transposed; K-matrix and V^T naturally in this layout]
//   out:   Cf (f32, optional accumulate) or Ch (f16), optional transposed
//          store (used for V^T), optional f32 bias[n].
// Block tile 128x64, K-step 32, 256 threads = 8 wave32 waves; each wave owns
// a 32x32 patch = 4 WMMA tiles. Double-buffered LDS, async staging.
// ---------------------------------------------------------------------------
template<bool FULL>
__device__ __forceinline__
void gemm_body(const _Float16* __restrict__ Ap, int lda,
               const _Float16* __restrict__ Bp, int ldb,
               float* __restrict__ Cf, _Float16* __restrict__ Ch,
               int ldc, int storeT,
               int M, int N, int K,
               const float* __restrict__ bias, int accum,
               int m0, int n0,
               _Float16 (&As)[2][128][40], _Float16 (&Bs)[2][64][40])
{
    const int tid  = threadIdx.x;
    const int lane = tid & 31;
    const int wave = tid >> 5;
    const int wm = (wave & 3) * 32;   // wave row base in tile
    const int wn = (wave >> 2) * 32;  // wave col base in tile

    v8f c00 = {}; v8f c01 = {}; v8f c10 = {}; v8f c11 = {};

    // staging split (256 threads): A: 128 rows x 32 halves, B: 64 rows x 32
    const int ar  = tid >> 1;         // 0..127
    const int ac0 = (tid & 1) * 16;   // 0/16
    const int bn  = tid & 63;         // 0..63
    const int bkr = (tid >> 6) * 8;   // 0/8/16/24

    if (FULL) {
        // prologue: stage K-tile 0 into buffer 0 (3 async b128 per thread)
        const _Float16* ga = Ap + (size_t)(m0 + ar) * lda + ac0;
        async_cp16(ga,     &As[0][ar][ac0]);
        async_cp16(ga + 8, &As[0][ar][ac0 + 8]);
        async_cp16(Bp + (size_t)(n0 + bn) * ldb + bkr, &Bs[0][bn][bkr]);
    }

    for (int k0 = 0; k0 < K; k0 += 32) {
        const int  buf  = (k0 >> 5) & 1;
        const bool more = (k0 + 32) < K;

        if (FULL) {
            if (more) {   // stage next K-tile into the other buffer
                const int kn = k0 + 32;
                const _Float16* ga = Ap + (size_t)(m0 + ar) * lda + kn + ac0;
                async_cp16(ga,     &As[buf ^ 1][ar][ac0]);
                async_cp16(ga + 8, &As[buf ^ 1][ar][ac0 + 8]);
                async_cp16(Bp + (size_t)(n0 + bn) * ldb + kn + bkr,
                           &Bs[buf ^ 1][bn][bkr]);
                wait_async_3();    // current buffer's 3 oldest are done
            } else {
                wait_async_0();
            }
        } else {
            // guarded (ragged) path: synchronous bounded copies into buffer
            const int gm = m0 + ar;
            #pragma unroll
            for (int i = 0; i < 16; ++i) {
                const int gk = k0 + ac0 + i;
                As[buf][ar][ac0 + i] = (gm < M && gk < K)
                    ? Ap[(size_t)gm * lda + gk] : (_Float16)0.0f;
            }
            const int gn = n0 + bn;
            #pragma unroll
            for (int i = 0; i < 8; ++i) {
                const int gk = k0 + bkr + i;
                Bs[buf][bn][bkr + i] = (gn < N && gk < K)
                    ? Bp[(size_t)gn * ldb + gk] : (_Float16)0.0f;
            }
        }
        __syncthreads();

        // ---- fragments (ISA 7.12.2 layouts), contiguous b128 LDS reads ----
        _Float16 (&Ab)[128][40] = As[buf];
        _Float16 (&Bb)[64][40]  = Bs[buf];

        Frag16 a0, a1, b0, b1;
        const int am = wm + (lane & 15);
        const int kb = (lane >> 4) * 8;     // A: lanes 0-15 K{0-7,16-23}, hi +8
        a0.h[0] = *(const v8h*)&Ab[am][kb];
        a0.h[1] = *(const v8h*)&Ab[am][kb + 16];
        a1.h[0] = *(const v8h*)&Ab[am + 16][kb];
        a1.h[1] = *(const v8h*)&Ab[am + 16][kb + 16];

        const int bc  = lane & 15;
        const int bkb = (lane >> 4) * 8;    // B: lanes 0-15 K{0-7,16-23}, hi +8
        b0.h[0] = *(const v8h*)&Bb[wn + bc][bkb];
        b0.h[1] = *(const v8h*)&Bb[wn + bc][bkb + 16];
        b1.h[0] = *(const v8h*)&Bb[wn + 16 + bc][bkb];
        b1.h[1] = *(const v8h*)&Bb[wn + 16 + bc][bkb + 16];

        c00 = __builtin_amdgcn_wmma_f32_16x16x32_f16(false, a0.v, false, b0.v,
                                                     (short)0, c00, false, false);
        c01 = __builtin_amdgcn_wmma_f32_16x16x32_f16(false, a0.v, false, b1.v,
                                                     (short)0, c01, false, false);
        c10 = __builtin_amdgcn_wmma_f32_16x16x32_f16(false, a1.v, false, b0.v,
                                                     (short)0, c10, false, false);
        c11 = __builtin_amdgcn_wmma_f32_16x16x32_f16(false, a1.v, false, b1.v,
                                                     (short)0, c11, false, false);
        __syncthreads();   // all waves done reading buf before it is re-staged
    }

    // ---- epilogue (C/D layout: VGPR r -> M = r + 8*(lane>=16), N = lane&15)
    const int sm = (lane >> 4) * 8;
    const int sn = lane & 15;

    auto emit = [&](int gm, int gn, float val) {
        if (!FULL && (gm >= M || gn >= N)) return;
        float v = val;
        if (bias) v += bias[gn];
        const size_t idx = storeT ? (size_t)gn * ldc + gm
                                  : (size_t)gm * ldc + gn;
        if (Cf) {
            if (accum) v += Cf[idx];
            Cf[idx] = v;
        } else {
            Ch[idx] = (_Float16)v;
        }
    };

    #pragma unroll
    for (int r = 0; r < 8; ++r) {
        const int gm0 = m0 + wm + sm + r;
        const int gm1 = gm0 + 16;
        const int gn0 = n0 + wn + sn;
        const int gn1 = gn0 + 16;
        emit(gm0, gn0, c00[r]);
        emit(gm0, gn1, c01[r]);
        emit(gm1, gn0, c10[r]);
        emit(gm1, gn1, c11[r]);
    }
}

__global__ __launch_bounds__(256)
void gemm_wmma_kernel(const _Float16* __restrict__ Ap, int lda,
                      const _Float16* __restrict__ Bp, int ldb,
                      float* __restrict__ Cf, _Float16* __restrict__ Ch,
                      int ldc, int storeT,
                      int M, int N, int K,
                      const float* __restrict__ bias, int accum)
{
    __shared__ __align__(16) _Float16 As[2][128][40];
    __shared__ __align__(16) _Float16 Bs[2][64][40];
    const int m0 = blockIdx.y * 128;
    const int n0 = blockIdx.x * 64;
    const bool full = (m0 + 128 <= M) && (n0 + 64 <= N) && ((K & 31) == 0) &&
                      ((lda & 7) == 0) && ((ldb & 7) == 0);
    if (full)
        gemm_body<true >(Ap, lda, Bp, ldb, Cf, Ch, ldc, storeT,
                         M, N, K, bias, accum, m0, n0, As, Bs);
    else
        gemm_body<false>(Ap, lda, Bp, ldb, Cf, Ch, ldc, storeT,
                         M, N, K, bias, accum, m0, n0, As, Bs);
}

// ---------------------------------------------------------------------------
// Convert f32 [batch][R][C] -> f16 transposed [batch][C][R] (weights -> n-major)
// ---------------------------------------------------------------------------
__global__ __launch_bounds__(256)
void cvtT_kernel(const float* __restrict__ src, _Float16* __restrict__ dst,
                 int R, int C)
{
    const size_t base = (size_t)blockIdx.z * R * C;
    const int r = blockIdx.y;
    for (int c = blockIdx.x * 256 + threadIdx.x; c < C; c += gridDim.x * 256)
        dst[base + (size_t)c * R + r] = (_Float16)src[base + (size_t)r * C + c];
}

// ---------------------------------------------------------------------------
// Embedding + sinusoidal positional encoding: x[b,s,:] = emb[id] + PE[s,:]
// ---------------------------------------------------------------------------
__global__ __launch_bounds__(256)
void embed_kernel(const int* __restrict__ ids, const float* __restrict__ emb,
                  float* __restrict__ x)
{
    const int bs = blockIdx.x;
    const int s  = bs % S_;
    const int id = ids[bs];
    const float neg_log = -logf(10000.0f) / (float)E_;
    for (int e = threadIdx.x; e < E_; e += 256) {
        const int   i2  = e & ~1;
        const float ang = (float)s * expf((float)i2 * neg_log);
        const float pe  = (e & 1) ? cosf(ang) : sinf(ang);
        x[(size_t)bs * E_ + e] = emb[(size_t)id * E_ + e] + pe;
    }
}

// ---------------------------------------------------------------------------
// LayerNorm over last dim n, f32 in -> f16 out (one block per row)
// ---------------------------------------------------------------------------
__global__ __launch_bounds__(256)
void ln_kernel(const float* __restrict__ x, const float* __restrict__ g,
               const float* __restrict__ b, _Float16* __restrict__ y, int n)
{
    __shared__ float red[256];
    const int tid = threadIdx.x;
    const float* xr = x + (size_t)blockIdx.x * n;

    float s = 0.0f;
    for (int i = tid; i < n; i += 256) s += xr[i];
    red[tid] = s; __syncthreads();
    for (int o = 128; o > 0; o >>= 1) {
        if (tid < o) red[tid] += red[tid + o];
        __syncthreads();
    }
    const float mean = red[0] / (float)n;
    __syncthreads();

    s = 0.0f;
    for (int i = tid; i < n; i += 256) { float d = xr[i] - mean; s += d * d; }
    red[tid] = s; __syncthreads();
    for (int o = 128; o > 0; o >>= 1) {
        if (tid < o) red[tid] += red[tid + o];
        __syncthreads();
    }
    const float inv = rsqrtf(red[0] / (float)n + EPS_);

    _Float16* yr = y + (size_t)blockIdx.x * n;
    for (int i = tid; i < n; i += 256)
        yr[i] = (_Float16)((xr[i] - mean) * inv * g[i] + b[i]);
}

// ---------------------------------------------------------------------------
// Row softmax: f32 scores in, f16 probabilities out (scale + causal + pad mask)
// ---------------------------------------------------------------------------
__global__ __launch_bounds__(256)
void softmax_kernel(const float* __restrict__ sc, _Float16* __restrict__ pr,
                    const int* __restrict__ amask, float scale)
{
    __shared__ float red[256];
    const int tid = threadIdx.x;
    const int row = blockIdx.x;
    const float* r = sc + (size_t)row * S_;
    _Float16*    p = pr + (size_t)row * S_;

    float mx = -INFINITY;
    for (int c = tid; c < S_; c += 256) {
        const bool valid = (c <= row) && (amask[c] != 0);
        if (valid) mx = fmaxf(mx, r[c] * scale);
    }
    red[tid] = mx; __syncthreads();
    for (int o = 128; o > 0; o >>= 1) {
        if (tid < o) red[tid] = fmaxf(red[tid], red[tid + o]);
        __syncthreads();
    }
    mx = red[0]; __syncthreads();

    float sum = 0.0f;
    for (int c = tid; c < S_; c += 256) {
        const bool valid = (c <= row) && (amask[c] != 0);
        const float v = valid ? expf(r[c] * scale - mx) : 0.0f;
        sum += v;
    }
    red[tid] = sum; __syncthreads();
    for (int o = 128; o > 0; o >>= 1) {
        if (tid < o) red[tid] += red[tid + o];
        __syncthreads();
    }
    const float inv = 1.0f / red[0];
    for (int c = tid; c < S_; c += 256) {
        const bool valid = (c <= row) && (amask[c] != 0);
        const float v = valid ? expf(r[c] * scale - mx) * inv : 0.0f;
        p[c] = (_Float16)v;
    }
}

// ---------------------------------------------------------------------------
// Exact-erf GELU on f16 buffer (f32 math)
// ---------------------------------------------------------------------------
__global__ __launch_bounds__(256)
void gelu16_kernel(_Float16* __restrict__ x, size_t n)
{
    const size_t stride = (size_t)gridDim.x * 256;
    for (size_t i = (size_t)blockIdx.x * 256 + threadIdx.x; i < n; i += stride) {
        const float v = (float)x[i];
        x[i] = (_Float16)(0.5f * v * (1.0f + erff(v * 0.70710678118654752f)));
    }
}

// ---------------------------------------------------------------------------
// Host orchestration
// ---------------------------------------------------------------------------
extern "C" void kernel_launch(void* const* d_in, const int* in_sizes, int n_in,
                              void* d_out, int out_size, void* d_ws, size_t ws_size,
                              hipStream_t stream)
{
    (void)in_sizes; (void)n_in; (void)out_size; (void)ws_size;

    const int*   ids   = (const int*)  d_in[0];
    const int*   amask = (const int*)  d_in[1];
    const float* emb   = (const float*)d_in[2];
    const float* Wq    = (const float*)d_in[3];
    const float* Wk    = (const float*)d_in[4];
    const float* Wv    = (const float*)d_in[5];
    const float* Wo    = (const float*)d_in[6];
    const float* ln1g  = (const float*)d_in[7];
    const float* ln1b  = (const float*)d_in[8];
    const float* ln2g  = (const float*)d_in[9];
    const float* ln2b  = (const float*)d_in[10];
    const float* ffw1  = (const float*)d_in[11];
    const float* ffb1  = (const float*)d_in[12];
    const float* ffw2  = (const float*)d_in[13];
    const float* ffb2  = (const float*)d_in[14];
    const float* lnfg  = (const float*)d_in[15];
    const float* lnfb  = (const float*)d_in[16];
    const float* Wout  = (const float*)d_in[17];
    const float* bout  = (const float*)d_in[18];
    float*       out   = (float*)d_out;

    const int M = B_ * S_;

    // ---------------- workspace carve-up (~148 MB) ----------------
    char* wsb = (char*)d_ws;
    size_t off = 0;
    auto allocF = [&](size_t n) { float* p = (float*)(wsb + off);
                                  off += n * sizeof(float); return p; };
    auto allocH = [&](size_t n) { _Float16* p = (_Float16*)(wsb + off);
                                  off += n * sizeof(_Float16); return p; };

    float*    x      = allocF((size_t)M * E_);      // residual stream (f32)
    float*    scores = allocF((size_t)S_ * S_);     // attention scores (f32)
    _Float16* xn16   = allocH((size_t)M * E_);      // LN output
    _Float16* q16    = allocH((size_t)M * A_);
    _Float16* k16    = allocH((size_t)M * A_);
    _Float16* vT16   = allocH((size_t)M * A_);      // [h][d][b*S+s]
    _Float16* attn16 = allocH((size_t)M * A_);      // concat heads
    _Float16* probs16= allocH((size_t)S_ * S_);
    _Float16* hbuf16 = allocH((size_t)M * FF_);
    _Float16* WqT    = allocH((size_t)L_ * H_ * E_ * HD_);  // [l][h][d][e]
    _Float16* WkT    = allocH((size_t)L_ * H_ * E_ * HD_);
    _Float16* WvT    = allocH((size_t)L_ * H_ * E_ * HD_);
    _Float16* WoT    = allocH((size_t)L_ * A_ * E_);        // [l][e][a]
    _Float16* W1T    = allocH((size_t)L_ * E_ * FF_);       // [l][ff][e]
    _Float16* W2T    = allocH((size_t)L_ * FF_ * E_);       // [l][e][ff]
    _Float16* WoutT  = allocH((size_t)E_ * V_);             // [v][e]

    // ---------------- weight conversion + transpose (once) ----------------
    auto cvtT = [&](const float* src, _Float16* dst, int R, int C, int batch) {
        dim3 g((C + 255) / 256, R, batch);
        cvtT_kernel<<<g, 256, 0, stream>>>(src, dst, R, C);
    };
    cvtT(Wq,   WqT,   E_,  HD_, L_ * H_);
    cvtT(Wk,   WkT,   E_,  HD_, L_ * H_);
    cvtT(Wv,   WvT,   E_,  HD_, L_ * H_);
    cvtT(Wo,   WoT,   A_,  E_,  L_);
    cvtT(ffw1, W1T,   E_,  FF_, L_);
    cvtT(ffw2, W2T,   FF_, E_,  L_);
    cvtT(Wout, WoutT, E_,  V_,  1);

    // ---------------- GEMM wrapper ----------------
    auto gemm = [&](const _Float16* A, int lda, const _Float16* B, int ldb,
                    float* Cf, _Float16* Ch, int ldc, int storeT,
                    int m, int n, int k, const float* bias, int acc) {
        dim3 g((n + 63) / 64, (m + 127) / 128);
        gemm_wmma_kernel<<<g, 256, 0, stream>>>(A, lda, B, ldb, Cf, Ch, ldc,
                                                storeT, m, n, k, bias, acc);
    };

    const float scale = 0.125f;   // 1/sqrt(HD)

    embed_kernel<<<M, 256, 0, stream>>>(ids, emb, x);

    for (int l = 0; l < L_; ++l) {
        // --- attention ---
        ln_kernel<<<M, 256, 0, stream>>>(x, ln1g + (size_t)l * E_,
                                         ln1b + (size_t)l * E_, xn16, E_);
        for (int h = 0; h < H_; ++h) {
            const size_t wofs = ((size_t)l * H_ + h) * E_ * HD_;
            gemm(xn16, E_, WqT + wofs, E_, nullptr, q16 + h * HD_, A_, 0,
                 M, HD_, E_, nullptr, 0);
            gemm(xn16, E_, WkT + wofs, E_, nullptr, k16 + h * HD_, A_, 0,
                 M, HD_, E_, nullptr, 0);
            // V stored transposed: vT[h][d][b*S+s]
            gemm(xn16, E_, WvT + wofs, E_, nullptr, vT16 + (size_t)h * HD_ * M,
                 M, 1, M, HD_, E_, nullptr, 0);
        }
        for (int b = 0; b < B_; ++b) {
            for (int h = 0; h < H_; ++h) {
                const size_t qk = (size_t)b * S_ * A_ + (size_t)h * HD_;
                // scores = Q * K^T  (k16 rows are n-major over t, contiguous d)
                gemm(q16 + qk, A_, k16 + qk, A_, scores, nullptr, S_, 0,
                     S_, S_, HD_, nullptr, 0);
                softmax_kernel<<<S_, 256, 0, stream>>>(scores, probs16,
                                                       amask + (size_t)b * S_,
                                                       scale);
                // attn = P * V   (V^T rows are n-major over d, contiguous t)
                gemm(probs16, S_, vT16 + (size_t)h * HD_ * M + (size_t)b * S_,
                     M, nullptr, attn16 + qk, A_, 0, S_, HD_, S_, nullptr, 0);
            }
        }
        // x += attn * Wo[l]
        gemm(attn16, A_, WoT + (size_t)l * E_ * A_, A_, x, nullptr, E_, 0,
             M, E_, A_, nullptr, 1);

        // --- FFN ---
        ln_kernel<<<M, 256, 0, stream>>>(x, ln2g + (size_t)l * E_,
                                         ln2b + (size_t)l * E_, xn16, E_);
        gemm(xn16, E_, W1T + (size_t)l * FF_ * E_, E_, nullptr, hbuf16, FF_, 0,
             M, FF_, E_, ffb1 + (size_t)l * FF_, 0);
        gelu16_kernel<<<2048, 256, 0, stream>>>(hbuf16, (size_t)M * FF_);
        gemm(hbuf16, FF_, W2T + (size_t)l * E_ * FF_, FF_, x, nullptr, E_, 0,
             M, E_, FF_, ffb2 + (size_t)l * E_, 1);
    }

    // --- final LN + LM head ---
    ln_kernel<<<M, 256, 0, stream>>>(x, lnfg, lnfb, xn16, E_);
    gemm(xn16, E_, WoutT, E_, out, nullptr, V_, 0, M, V_, E_, bout, 0);
}
